// Qwen3Attention_54382875902552
// MI455X (gfx1250) — matrix operand order repaired and checked by reading, not compiled, not run on hardware
//
#include <hip/hip_runtime.h>
#include <hip/hip_bf16.h>

typedef __bf16 bf16;
typedef __attribute__((ext_vector_type(8)))  __bf16 bf16x8;
typedef __attribute__((ext_vector_type(16))) __bf16 bf16x16;
typedef __attribute__((ext_vector_type(8)))  float  v8f;

#define SEQ     2048
#define HIDDEN  4096
#define NH      32
#define NKV     8
#define HD      128
#define QKV_OUT 6144   // (NH + 2*NKV) * HD
#define O_IN    4096   // NH * HD

// ---------------------------------------------------------------------------
// WMMA helpers (CDNA5 16x16x32 bf16, f32 accumulate)
// ---------------------------------------------------------------------------
static __device__ __forceinline__ v8f wmma_bf16(bf16x16 a, bf16x16 b, v8f c) {
  return __builtin_amdgcn_wmma_f32_16x16x32_bf16(
      /*neg_a=*/false, a, /*neg_b=*/false, b,
      /*c_mod=*/(short)0, c, /*reuse_a=*/false, /*reuse_b=*/false);
}

// B-operand load: 16 contiguous bf16 (lane supplies hi*16 K-offset itself).
static __device__ __forceinline__ bf16x16 ld16(const bf16* p) {
  bf16x8 a = *(const bf16x8*)(p);
  bf16x8 b = *(const bf16x8*)(p + 8);
  bf16x16 r;
#pragma unroll
  for (int i = 0; i < 8; ++i) { r[i] = a[i]; r[i + 8] = b[i]; }
  return r;
}

// A-operand load: row base at k0; chunks at k=hi*16 and k=hi*16+8 (contiguous).
static __device__ __forceinline__ bf16x16 ldA(const bf16* rowbase, int hi) {
  bf16x8 a = *(const bf16x8*)(rowbase + hi * 16);
  bf16x8 b = *(const bf16x8*)(rowbase + hi * 16 + 8);
  bf16x16 r;
#pragma unroll
  for (int i = 0; i < 8; ++i) { r[i] = a[i]; r[i + 8] = b[i]; }
  return r;
}

static __device__ __forceinline__ v8f v8f_zero() {
  v8f z;
#pragma unroll
  for (int i = 0; i < 8; ++i) z[i] = 0.0f;
  return z;
}

// CDNA5 async copy global -> LDS (ASYNCcnt-tracked, no VGPR staging).
static __device__ __forceinline__ void async_g2l_b128(unsigned lds_off,
                                                      const bf16* g) {
  asm volatile("global_load_async_to_lds_b128 %0, %1, off"
               :: "v"(lds_off), "v"((unsigned long long)(uintptr_t)g)
               : "memory");
}

// ---------------------------------------------------------------------------
// Precision converts
// ---------------------------------------------------------------------------
__global__ __launch_bounds__(256)
void f32_to_bf16_kernel(const float* __restrict__ in, bf16* __restrict__ out,
                        size_t n) {
  size_t i = (size_t)blockIdx.x * blockDim.x + threadIdx.x;
  if (i < n) out[i] = (bf16)in[i];
}

// in: f32 [R][C] row-major  ->  out: bf16 [C][R] row-major (i.e. B^T layout)
__global__ __launch_bounds__(256)
void transpose_f32_to_bf16(const float* __restrict__ in, bf16* __restrict__ out,
                           int R, int C) {
  __shared__ float tile[32][33];
  int c0 = blockIdx.x * 32;
  int r0 = blockIdx.y * 32;
#pragma unroll
  for (int i = threadIdx.y; i < 32; i += 8)
    tile[i][threadIdx.x] = in[(size_t)(r0 + i) * C + c0 + threadIdx.x];
  __syncthreads();
#pragma unroll
  for (int i = threadIdx.y; i < 32; i += 8)
    out[(size_t)(c0 + i) * R + r0 + threadIdx.x] = (bf16)tile[threadIdx.x][i];
}

// ---------------------------------------------------------------------------
// bf16 WMMA GEMM:  C[M][N] f32 = A[M][K] bf16 (row major) * BT[N][K] bf16
// Block = 4 waves, tile 64(M) x 128(N); wave tile 64x32 (4x2 WMMA frags).
// A tile (64x64 per K-stage) is staged in LDS via global_load_async_to_lds,
// double-buffered: stage n+1 copies overlap stage n compute.
// ---------------------------------------------------------------------------
#define KSTEP 64
#define LDA   72   // padded LDS row stride (bf16 elems); 36*r mod 64 permutes banks

__global__ __launch_bounds__(128)
void gemm_bf16_kernel(const bf16* __restrict__ A, const bf16* __restrict__ BT,
                      float* __restrict__ C, int M, int N, int K) {
  __shared__ bf16 ldsA[2][64 * LDA];
  const int tid  = threadIdx.x;
  const int lane = tid & 31;
  const int wave = tid >> 5;
  const int lo   = lane & 15;
  const int hi   = lane >> 4;
  const int m0   = blockIdx.x * 64;
  const int n0   = blockIdx.y * 128 + wave * 32;

  v8f acc[4][2];
#pragma unroll
  for (int r = 0; r < 4; ++r)
#pragma unroll
    for (int c = 0; c < 2; ++c) acc[r][c] = v8f_zero();

  const bf16* bcol[2];
#pragma unroll
  for (int c = 0; c < 2; ++c)
    bcol[c] = BT + (size_t)(n0 + c * 16 + lo) * K + hi * 16;

  // Issue one K-stage of async A-tile copies: 512 x 16B segments, 4 per thread.
  auto issue = [&](int stage) {
    const int kk  = stage * KSTEP;
    const int buf = stage & 1;
#pragma unroll
    for (int i = 0; i < 4; ++i) {
      int s   = tid + i * 128;     // 0..511
      int row = s >> 3;            // 0..63
      int seg = s & 7;             // 8 bf16 (16B) each
      const bf16* g = A + (size_t)(m0 + row) * K + kk + seg * 8;
      unsigned lds_off = (unsigned)(uintptr_t)&ldsA[buf][row * LDA + seg * 8];
      async_g2l_b128(lds_off, g);
    }
  };

  const int ns = K / KSTEP;
  issue(0);
  if (ns > 1) issue(1);

  for (int n = 0; n < ns; ++n) {
    // Stage n done when at most the 4 next-stage copies remain outstanding.
    if (n + 1 < ns) asm volatile("s_wait_asynccnt 0x4" ::: "memory");
    else            asm volatile("s_wait_asynccnt 0x0" ::: "memory");
    __syncthreads();

    const bf16* abuf = &ldsA[n & 1][0];
    const int kk = n * KSTEP;
#pragma unroll
    for (int c2 = 0; c2 < 2; ++c2) {   // two K=32 chunks per stage
      bf16x16 a[4], b[2];
#pragma unroll
      for (int r = 0; r < 4; ++r)
        a[r] = ldA(abuf + (r * 16 + lo) * LDA + c2 * 32, hi);  // ds_load_b128 x2
#pragma unroll
      for (int c = 0; c < 2; ++c) b[c] = ld16(bcol[c] + kk + c2 * 32);
#pragma unroll
      for (int r = 0; r < 4; ++r)
#pragma unroll
        for (int c = 0; c < 2; ++c) acc[r][c] = wmma_bf16(a[r], b[c], acc[r][c]);
    }

    __syncthreads();                 // everyone done with buf before refill
    if (n + 2 < ns) issue(n + 2);
  }

#pragma unroll
  for (int r = 0; r < 4; ++r)
#pragma unroll
    for (int c = 0; c < 2; ++c)
#pragma unroll
      for (int vr = 0; vr < 8; ++vr)
        C[(size_t)(m0 + r * 16 + vr + 8 * hi) * N + (n0 + c * 16 + lo)] =
            acc[r][c][vr];
}

// ---------------------------------------------------------------------------
// Fused per-head RMSNorm + RoPE; emits bf16 Q [h][t][d], K [kv][t][d],
// V transposed [kv][d][t] (B layout for the P*V WMMA).
// One block per token t; one wave per 128-wide head slice, 4 elems/lane.
// ---------------------------------------------------------------------------
__global__ __launch_bounds__(256)
void normrope_kernel(const float* __restrict__ qkv, const float* __restrict__ cosT,
                     const float* __restrict__ sinT, const float* __restrict__ qw,
                     const float* __restrict__ kw, bf16* __restrict__ Q,
                     bf16* __restrict__ Kb, bf16* __restrict__ Vt) {
  const int t    = blockIdx.x;
  const int lane = threadIdx.x & 31;
  const int wave = threadIdx.x >> 5;
  const float* row  = qkv  + (size_t)t * QKV_OUT;
  const float* crow = cosT + (size_t)t * 64;
  const float* srow = sinT + (size_t)t * 64;

  for (int s = wave; s < 48; s += 8) {
    const float4 x4 = *(const float4*)(row + s * HD + lane * 4);
    float x[4] = {x4.x, x4.y, x4.z, x4.w};

    if (s >= 40) {  // V: passthrough, store transposed [kv][d][t]
      int h = s - 40;
#pragma unroll
      for (int j = 0; j < 4; ++j) {
        int d = lane * 4 + j;
        Vt[((size_t)h * HD + d) * SEQ + t] = (bf16)x[j];
      }
      continue;
    }

    // RMS norm over 128 dims (full-wave reduction)
    float ss = x[0] * x[0] + x[1] * x[1] + x[2] * x[2] + x[3] * x[3];
#pragma unroll
    for (int m = 1; m <= 16; m <<= 1) ss += __shfl_xor(ss, m);
    float rms = rsqrtf(ss * (1.0f / 128.0f) + 1e-6f);
    const float* w = (s < 32) ? qw : kw;

    float xn[4];
#pragma unroll
    for (int j = 0; j < 4; ++j) xn[j] = x[j] * rms * w[lane * 4 + j];

    // RoPE: pair d <-> d+64 lives in lane^16, same slot.
    float px[4];
#pragma unroll
    for (int j = 0; j < 4; ++j) px[j] = __shfl_xor(xn[j], 16);

    float out[4];
#pragma unroll
    for (int j = 0; j < 4; ++j) {
      int d  = lane * 4 + j;
      int j2 = d & 63;
      float c = crow[j2], sn = srow[j2];
      out[j] = (d < 64) ? (xn[j] * c - px[j] * sn)   // x1*c - x2*s
                        : (xn[j] * c + px[j] * sn);  // x2*c + x1*s
    }

    if (s < 32) {
      bf16* dst = Q + ((size_t)s * SEQ + t) * HD + lane * 4;
#pragma unroll
      for (int j = 0; j < 4; ++j) dst[j] = (bf16)out[j];
    } else {
      int h = s - 32;
      bf16* dst = Kb + ((size_t)h * SEQ + t) * HD + lane * 4;
#pragma unroll
      for (int j = 0; j < 4; ++j) dst[j] = (bf16)out[j];
    }
  }
}

// ---------------------------------------------------------------------------
// Causal flash attention, GQA 32q/8kv heads, head_dim 128.
// One wave per 16-query tile; key blocks of 32; online softmax in f32.
// Per key block: 8 score WMMAs + 8 P*V WMMAs; P transposed C->A layout via LDS.
// ---------------------------------------------------------------------------
__global__ __launch_bounds__(128)
void attn_kernel(const bf16* __restrict__ Q, const bf16* __restrict__ Kb,
                 const bf16* __restrict__ Vt, bf16* __restrict__ O) {
  __shared__ float sP[4][16][33];  // per-wave 16x32 P tile (padded)
  const int lane = threadIdx.x & 31;
  const int wave = threadIdx.x >> 5;
  const int lo   = lane & 15;
  const int hi   = lane >> 4;
  const int head = blockIdx.y;
  const int kv   = head >> 2;              // 4 q heads per kv head
  const int q0   = (blockIdx.x * 4 + wave) * 16;
  const float scale = 0.08838834764831845f;  // 1/sqrt(128)

  const bf16* qbase = Q  + ((size_t)head * SEQ + q0) * HD;
  const bf16* kbase = Kb + (size_t)kv * SEQ * HD;
  const bf16* vbase = Vt + (size_t)kv * HD * SEQ;

  // Q tile in A layout: 4 chunks of K=32 over d=0..127
  bf16x16 qa[4];
#pragma unroll
  for (int ch = 0; ch < 4; ++ch)
    qa[ch] = ldA(qbase + (size_t)lo * HD + ch * 32, hi);

  v8f o[8];
#pragma unroll
  for (int dt = 0; dt < 8; ++dt) o[dt] = v8f_zero();
  float mrow[8], lrow[8];
#pragma unroll
  for (int vr = 0; vr < 8; ++vr) { mrow[vr] = -INFINITY; lrow[vr] = 0.0f; }

  float(*myP)[33] = sP[wave];

  for (int k0 = 0; k0 <= q0 + 15; k0 += 32) {
    v8f s0 = v8f_zero(), s1 = v8f_zero();
#pragma unroll
    for (int ch = 0; ch < 4; ++ch) {
      s0 = wmma_bf16(qa[ch],
                     ld16(kbase + (size_t)(k0 + lo) * HD + ch * 32 + hi * 16), s0);
      s1 = wmma_bf16(qa[ch],
                     ld16(kbase + (size_t)(k0 + 16 + lo) * HD + ch * 32 + hi * 16), s1);
    }

    const bool msk = (k0 + 31 > q0);
    float alpha[8];
#pragma unroll
    for (int vr = 0; vr < 8; ++vr) {
      float a = s0[vr] * scale, b = s1[vr] * scale;
      if (msk) {
        int qr = q0 + vr + 8 * hi;
        if (k0 + lo > qr)       a = -1e30f;
        if (k0 + 16 + lo > qr)  b = -1e30f;
      }
      float tmax = fmaxf(a, b);
#pragma unroll
      for (int m = 1; m <= 8; m <<= 1) tmax = fmaxf(tmax, __shfl_xor(tmax, m));
      float mnew = fmaxf(mrow[vr], tmax);
      alpha[vr]  = __expf(mrow[vr] - mnew);
      float p0 = __expf(a - mnew);
      float p1 = __expf(b - mnew);
      float rs = p0 + p1;
#pragma unroll
      for (int m = 1; m <= 8; m <<= 1) rs += __shfl_xor(rs, m);
      lrow[vr] = lrow[vr] * alpha[vr] + rs;
      mrow[vr] = mnew;
      // stash P in C layout (row = vr + 8*hi, cols lo / lo+16)
      myP[vr + 8 * hi][lo]      = p0;
      myP[vr + 8 * hi][16 + lo] = p1;
    }

#pragma unroll
    for (int dt = 0; dt < 8; ++dt)
#pragma unroll
      for (int vr = 0; vr < 8; ++vr) o[dt][vr] *= alpha[vr];

    // Re-read P in A layout (row = lo, K chunks hi*16 and hi*16+8), cvt bf16
    bf16x16 pa;
#pragma unroll
    for (int i = 0; i < 8; ++i) {
      pa[i]     = (bf16)myP[lo][hi * 16 + i];
      pa[i + 8] = (bf16)myP[lo][hi * 16 + 8 + i];
    }

    // out += P (16x32) * V (32x16 per d-tile); V stored [d][t]
#pragma unroll
    for (int dt = 0; dt < 8; ++dt)
      o[dt] = wmma_bf16(pa,
                        ld16(vbase + (size_t)(dt * 16 + lo) * SEQ + k0 + hi * 16),
                        o[dt]);
  }

  float inv[8];
#pragma unroll
  for (int vr = 0; vr < 8; ++vr) inv[vr] = 1.0f / lrow[vr];
#pragma unroll
  for (int dt = 0; dt < 8; ++dt)
#pragma unroll
    for (int vr = 0; vr < 8; ++vr)
      O[(size_t)(q0 + vr + 8 * hi) * O_IN + head * HD + dt * 16 + lo] =
          (bf16)(o[dt][vr] * inv[vr]);
}

// ---------------------------------------------------------------------------
// Launch
// ---------------------------------------------------------------------------
extern "C" void kernel_launch(void* const* d_in, const int* in_sizes, int n_in,
                              void* d_out, int out_size, void* d_ws, size_t ws_size,
                              hipStream_t stream) {
  (void)in_sizes; (void)n_in; (void)out_size; (void)ws_size;
  const float* hidden = (const float*)d_in[0];
  const float* cosT   = (const float*)d_in[1];
  const float* sinT   = (const float*)d_in[2];
  const float* w_qkv  = (const float*)d_in[3];
  const float* w_o    = (const float*)d_in[4];
  const float* qw     = (const float*)d_in[5];
  const float* kw     = (const float*)d_in[6];

  char* ws = (char*)d_ws;
  size_t off = 0;
  auto take = [&](size_t bytes) -> char* {
    char* p = ws + off;
    off += (bytes + 255) & ~(size_t)255;
    return p;
  };
  bf16*  hB    = (bf16*) take((size_t)SEQ * HIDDEN * 2);
  bf16*  wqkvT = (bf16*) take((size_t)QKV_OUT * HIDDEN * 2);
  bf16*  woT   = (bf16*) take((size_t)HIDDEN * O_IN * 2);
  float* qkv   = (float*)take((size_t)SEQ * QKV_OUT * 4);
  bf16*  Qb    = (bf16*) take((size_t)NH * SEQ * HD * 2);
  bf16*  Kbuf  = (bf16*) take((size_t)NKV * SEQ * HD * 2);
  bf16*  Vtb   = (bf16*) take((size_t)NKV * HD * SEQ * 2);
  bf16*  Ob    = (bf16*) take((size_t)SEQ * NH * HD * 2);

  // 1) precision converts (+ weight transposes into WMMA B layout)
  f32_to_bf16_kernel<<<(SEQ * HIDDEN) / 256, 256, 0, stream>>>(
      hidden, hB, (size_t)SEQ * HIDDEN);
  transpose_f32_to_bf16<<<dim3(QKV_OUT / 32, HIDDEN / 32), dim3(32, 8), 0, stream>>>(
      w_qkv, wqkvT, HIDDEN, QKV_OUT);
  transpose_f32_to_bf16<<<dim3(HIDDEN / 32, O_IN / 32), dim3(32, 8), 0, stream>>>(
      w_o, woT, O_IN, HIDDEN);

  // 2) QKV projection GEMM (2048x4096 @ 4096x6144)
  gemm_bf16_kernel<<<dim3(SEQ / 64, QKV_OUT / 128), 128, 0, stream>>>(
      hB, wqkvT, qkv, SEQ, QKV_OUT, HIDDEN);

  // 3) fused RMSNorm + RoPE, layout for attention
  normrope_kernel<<<SEQ, 256, 0, stream>>>(qkv, cosT, sinT, qw, kw, Qb, Kbuf, Vtb);

  // 4) causal flash attention (32 heads x 128 query tiles)
  attn_kernel<<<dim3(SEQ / 64, NH), 128, 0, stream>>>(Qb, Kbuf, Vtb, Ob);

  // 5) output projection GEMM (2048x4096 @ 4096x4096) -> f32 d_out
  gemm_bf16_kernel<<<dim3(SEQ / 64, HIDDEN / 128), 128, 0, stream>>>(
      Ob, woT, (float*)d_out, SEQ, HIDDEN, O_IN);
}
